// RNNLayer_89111981457822
// MI455X (gfx1250) — compile-verified, matrix operand order checked by reading, not run on hardware
//
#include <hip/hip_runtime.h>

// ---------------------------------------------------------------------------
// RNN layer for MI455X (gfx1250, wave32, WMMA, async global->LDS copies).
//   Kernel 0 (x2): W (fp32 row-major [k][n]) -> bf16 transposed [n][k] in ws.
//   Phase 1: xin = x_seq @ W_in + b_in  (bf16 WMMA GEMM; B tiles staged with
//            GLOBAL_LOAD_ASYNC_TO_LDS_B128, double-buffered LDS).
//   Phase 2: h_t = relu(xin_t + h_{t-1} @ W_hh) persistent kernel; W_hh slice
//            async-copied into LDS once; grid barrier per timestep.
// ---------------------------------------------------------------------------

typedef __attribute__((ext_vector_type(16))) __bf16 v16bf;
typedef __attribute__((ext_vector_type(8)))  __bf16 v8bf;
typedef __attribute__((ext_vector_type(8)))  float  v8f;
typedef __attribute__((ext_vector_type(4)))  int    v4i;

union BF16Tile { v16bf v; v8bf h[2]; };

constexpr int kB    = 32;
constexpr int kS    = 1024;
constexpr int kDin  = 1024;
constexpr int kDout = 1024;
constexpr int kM1   = kB * kS;          // 32768 rows in phase-1 GEMM

#define GAS __attribute__((address_space(1)))
#define LAS __attribute__((address_space(3)))

__device__ __forceinline__ unsigned short f2bf(float f) {
    unsigned int u = __float_as_uint(f);
    u += 0x7FFFu + ((u >> 16) & 1u);    // round-to-nearest-even
    return (unsigned short)(u >> 16);
}

// Async DMA: 16B global -> LDS, tracked by ASYNCcnt (no VGPR data path).
__device__ __forceinline__ void async_b128(void* lds_dst, const void* gsrc) {
    __builtin_amdgcn_global_load_async_to_lds_b128(
        (GAS v4i*)gsrc, (LAS v4i*)lds_dst, 0, 0);
}
__device__ __forceinline__ void wait_async0() {
    __builtin_amdgcn_s_wait_asynccnt(0);
}

// ---------------------------------------------------------------------------
// Kernel 0: W[1024][1024] fp32 (k,n) -> WT[1024][1024] bf16 (n,k).
// Coalesced float4 reads; scattered 2B writes (one-time, 4MB total).
// ---------------------------------------------------------------------------
__global__ __launch_bounds__(256) void wtrans_bf16_kernel(
    const float* __restrict__ W, unsigned short* __restrict__ WT)
{
    const int t  = blockIdx.x * 256 + threadIdx.x;  // 0..262143
    const int k  = t >> 8;                          // 0..1023
    const int n0 = (t & 255) * 4;                   // 0..1020
    const float4 v = *(const float4*)&W[(size_t)k * kDout + n0];
    WT[(size_t)(n0 + 0) * kDin + k] = f2bf(v.x);
    WT[(size_t)(n0 + 1) * kDin + k] = f2bf(v.y);
    WT[(size_t)(n0 + 2) * kDin + k] = f2bf(v.z);
    WT[(size_t)(n0 + 3) * kDin + k] = f2bf(v.w);
}

// ---------------------------------------------------------------------------
// Phase 1: C[M1,Dout] = X[M1,Din] * Win[Din,Dout] + bias
// 128x128 block tile, 8 waves (4x2 WMMA accs each), K=32 per step.
// A: fp32->bf16 VALU staging (read-once data, conversion unavoidable).
// B: async global->LDS from bf16 column-major WinT, double-buffered.
// ---------------------------------------------------------------------------
#define TM 128
#define TN 128
#define TK 32
#define NK (kDin / TK)

__global__ __launch_bounds__(256) void xin_gemm_kernel(
    const float* __restrict__ X,
    const unsigned short* __restrict__ WinT,   // [n][k] bf16
    const float* __restrict__ bin,
    float* __restrict__ Out)
{
    __shared__ unsigned short As[2][TM][TK];   // [m][k]   2 x 8 KB
    __shared__ unsigned short Bs[2][TN][TK];   // [n][k]   2 x 8 KB

    const int tid  = threadIdx.x;
    const int lane = tid & 31;
    const int wave = tid >> 5;
    const int l16  = lane & 15;
    const int half = lane >> 4;

    const int mBase = (blockIdx.x >> 3) * TM;
    const int nBase = (blockIdx.x & 7) * TN;

    const int wm = wave & 1;
    const int wn = wave >> 1;

    auto stageA = [&](int p, int kt) {
        const int k0 = kt * TK;
        #pragma unroll
        for (int q4 = 0; q4 < 4; ++q4) {
            const int q   = tid + q4 * 256;
            const int row = q >> 3;
            const int kq  = (q & 7) * 4;
            const float4 v = *(const float4*)&X[(size_t)(mBase + row) * kDin + k0 + kq];
            As[p][row][kq + 0] = f2bf(v.x);
            As[p][row][kq + 1] = f2bf(v.y);
            As[p][row][kq + 2] = f2bf(v.z);
            As[p][row][kq + 3] = f2bf(v.w);
        }
    };
    auto stageB = [&](int p, int kt) {
        const int k0 = kt * TK;
        #pragma unroll
        for (int q2 = 0; q2 < 2; ++q2) {
            const int c  = tid + q2 * 256;     // 0..511 16B chunks
            const int n  = c >> 2;             // 0..127
            const int kc = (c & 3) * 8;        // 0,8,16,24
            async_b128(&Bs[p][n][kc],
                       &WinT[(size_t)(nBase + n) * kDin + k0 + kc]);
        }
    };

    v8f acc[4][2];
    #pragma unroll
    for (int i = 0; i < 4; ++i)
        #pragma unroll
        for (int j = 0; j < 2; ++j) acc[i][j] = {};

    stageA(0, 0);
    stageB(0, 0);

    for (int kt = 0; kt < NK; ++kt) {
        const int p = kt & 1;
        wait_async0();                 // tile kt's async B copies done
        __syncthreads();               // publish A (ds) + B (async) for tile kt
        if (kt + 1 < NK) {             // prefetch next tile into other buffer,
            stageA(p ^ 1, kt + 1);     // overlapped with the WMMAs below
            stageB(p ^ 1, kt + 1);
        }

        BF16Tile b[2];
        #pragma unroll
        for (int j = 0; j < 2; ++j) {
            const int n0 = wn * 32 + j * 16 + l16;
            b[j].h[0] = *(const v8bf*)&Bs[p][n0][8 * half];
            b[j].h[1] = *(const v8bf*)&Bs[p][n0][16 + 8 * half];
        }
        #pragma unroll
        for (int i = 0; i < 4; ++i) {
            const int m0 = wm * 64 + i * 16 + l16;
            BF16Tile a;
            a.h[0] = *(const v8bf*)&As[p][m0][8 * half];
            a.h[1] = *(const v8bf*)&As[p][m0][16 + 8 * half];
            #pragma unroll
            for (int j = 0; j < 2; ++j)
                acc[i][j] = __builtin_amdgcn_wmma_f32_16x16x32_bf16(
                    false, a.v, false, b[j].v, (short)0, acc[i][j], false, false);
        }
        __syncthreads();               // all reads of buffer p done
    }

    #pragma unroll
    for (int i = 0; i < 4; ++i)
        #pragma unroll
        for (int j = 0; j < 2; ++j) {
            const int gcol = nBase + wn * 32 + j * 16 + l16;
            const float bias = bin[gcol];
            #pragma unroll
            for (int r = 0; r < 8; ++r) {
                const int grow = mBase + wm * 64 + i * 16 + r + 8 * half;
                Out[(size_t)grow * kDout + gcol] = acc[i][j][r] + bias;
            }
        }
}

// ---------------------------------------------------------------------------
// Phase 2: persistent scan, 16 WGs x 256 threads, WG owns 64 columns.
// W_hh slice async-copied from bf16 WhhT into LDS once (padded col stride).
// h kept as bf16 double buffer in global; grid barrier per timestep.
// ---------------------------------------------------------------------------
#define NWG2  16
#define WCOLS 64
#define KP    (kDout + 8)

__global__ __launch_bounds__(256) void rnn_scan_kernel(
    const unsigned short* __restrict__ WhhT,  // [n][k] bf16
    float* __restrict__ Out,                  // [B, S, Dout]: xin in, h out
    unsigned short* __restrict__ Hbuf,        // 2 * B * Dout bf16
    unsigned int* __restrict__ Cnt)           // [S] counters (pre-zeroed)
{
    extern __shared__ unsigned short Whs[];   // WCOLS * KP, Whs[n*KP + k]

    const int tid  = threadIdx.x;
    const int lane = tid & 31;
    const int wave = tid >> 5;
    const int l16  = lane & 15;
    const int half = lane >> 4;
    const int c0   = blockIdx.x * WCOLS;

    // one-time async DMA of the W_hh column slice into LDS (8192 x 16B)
    for (int c = tid; c < WCOLS * (kDout / 8); c += 256) {
        const int n  = c >> 7;           // 0..63
        const int kc = (c & 127) * 8;    // 0..1016
        async_b128(&Whs[n * KP + kc], &WhhT[(size_t)(c0 + n) * kDout + kc]);
    }
    wait_async0();
    __syncthreads();

    const int mt = wave & 1;
    const int nt = wave >> 1;
    const int n0 = nt * 16 + l16;
    const int m  = mt * 16 + l16;

    for (int t = 0; t < kS; ++t) {
        const unsigned short* Hc = Hbuf + (size_t)(t & 1) * (kB * kDout);
        unsigned short*       Hn = Hbuf + (size_t)((t + 1) & 1) * (kB * kDout);

        v8f acc = {};
        #pragma unroll 4
        for (int kt = 0; kt < kDout / 32; ++kt) {
            const int k0 = kt * 32;
            BF16Tile a, b;
            a.h[0] = *(const v8bf*)&Hc[(size_t)m * kDout + k0 + 8 * half];
            a.h[1] = *(const v8bf*)&Hc[(size_t)m * kDout + k0 + 16 + 8 * half];
            b.h[0] = *(const v8bf*)&Whs[n0 * KP + k0 + 8 * half];
            b.h[1] = *(const v8bf*)&Whs[n0 * KP + k0 + 16 + 8 * half];
            acc = __builtin_amdgcn_wmma_f32_16x16x32_bf16(
                false, a.v, false, b.v, (short)0, acc, false, false);
        }

        const int gcol = c0 + nt * 16 + l16;
        #pragma unroll
        for (int r = 0; r < 8; ++r) {
            const int brow = mt * 16 + r + 8 * half;
            const size_t oidx = ((size_t)brow * kS + t) * kDout + gcol;
            float v = acc[r] + Out[oidx];
            v = v > 0.0f ? v : 0.0f;
            Out[oidx] = v;
            Hn[(size_t)brow * kDout + gcol] = f2bf(v);
        }

        // grid-wide barrier for step t
        __threadfence();
        __syncthreads();
        if (tid == 0) {
            __hip_atomic_fetch_add(&Cnt[t], 1u, __ATOMIC_ACQ_REL, __HIP_MEMORY_SCOPE_AGENT);
            while (__hip_atomic_load(&Cnt[t], __ATOMIC_ACQUIRE, __HIP_MEMORY_SCOPE_AGENT) < NWG2)
                __builtin_amdgcn_s_sleep(1);
        }
        __syncthreads();
    }
}

// ---------------------------------------------------------------------------
extern "C" void kernel_launch(void* const* d_in, const int* in_sizes, int n_in,
                              void* d_out, int out_size, void* d_ws, size_t ws_size,
                              hipStream_t stream) {
    const float* x_seq = (const float*)d_in[0];   // [B, S, Din]
    const float* W_in  = (const float*)d_in[1];   // [Din, Dout]
    const float* b_in  = (const float*)d_in[2];   // [Dout]
    const float* W_hh  = (const float*)d_in[3];   // [Dout, Dout]
    float* out = (float*)d_out;                   // [B, S, Dout]

    // workspace layout (~4.2 MB)
    char* ws = (char*)d_ws;
    unsigned short* WinT = (unsigned short*)ws;                              // 2 MB
    unsigned short* WhhT = (unsigned short*)(ws + (size_t)2 * 1024 * 1024);  // 2 MB
    unsigned short* Hbuf = (unsigned short*)(ws + (size_t)4 * 1024 * 1024);  // 128 KB
    unsigned int*   Cnt  = (unsigned int*)(ws + (size_t)4 * 1024 * 1024
                                              + (size_t)2 * kB * kDout * sizeof(unsigned short));

    (void)hipMemsetAsync(Hbuf, 0, (size_t)2 * kB * kDout * sizeof(unsigned short), stream);
    (void)hipMemsetAsync(Cnt,  0, (size_t)kS * sizeof(unsigned int), stream);

    // Kernel 0: weight convert+transpose to bf16 column-major
    wtrans_bf16_kernel<<<dim3(1024), dim3(256), 0, stream>>>(W_in, WinT);
    wtrans_bf16_kernel<<<dim3(1024), dim3(256), 0, stream>>>(W_hh, WhhT);

    // Phase 1: 256 m-tiles * 8 n-tiles
    xin_gemm_kernel<<<dim3((kM1 / TM) * (kDout / TN)), dim3(256), 0, stream>>>(
        x_seq, WinT, b_in, out);

    // Phase 2: persistent, 16 co-resident WGs, ~129 KB dynamic LDS each
    rnn_scan_kernel<<<dim3(NWG2), dim3(256),
                      (size_t)WCOLS * KP * sizeof(unsigned short), stream>>>(
        WhhT, out, Hbuf, Cnt);
}
